// MoETransformerEncoderLayer_13606456394103
// MI455X (gfx1250) — compile-verified, matrix operand order checked by reading, not run on hardware
//
#include <hip/hip_runtime.h>
#include <hip/hip_bf16.h>
#include <math.h>
#include <stdint.h>

// MoE Transformer encoder layer for gfx1250 (MI455X), bf16 WMMA everywhere.
// B=2 S=1024 D=1024 H=16 HD=64 E=8 F=4096, T = B*S = 2048 tokens.

typedef __bf16 bf16_t;
typedef __attribute__((ext_vector_type(16))) __bf16 v16bf;
typedef __attribute__((ext_vector_type(8)))  float  v8f;
typedef __attribute__((ext_vector_type(4)))  float  v4f;

#define TOK 2048
#define DIM 1024
#define NH  16
#define HDIM 64
#define NE  8
#define FF  4096
#define SEQ 1024

// ---------------------------------------------------------------------------
// CDNA5 async helpers
// ---------------------------------------------------------------------------
template <int N>
__device__ __forceinline__ void wait_asynccnt() {
#if __has_builtin(__builtin_amdgcn_s_wait_asynccnt)
  __builtin_amdgcn_s_wait_asynccnt((short)N);
#else
  asm volatile("s_wait_asynccnt %0" ::"i"(N) : "memory");
#endif
}

// Per-lane 16B async copy global -> LDS (VFLAT async op, ASYNCcnt-tracked).
// lds_off is the byte offset within the workgroup LDS allocation (low 32 bits
// of the flat shared-pointer per the gfx1250 aperture mapping).
__device__ __forceinline__ void async_load_b128(uint32_t lds_off,
                                                const void* gptr) {
  asm volatile("global_load_async_to_lds_b128 %0, %1, off" ::"v"(lds_off),
               "v"((uint64_t)(uintptr_t)gptr)
               : "memory");
}

// ---------------------------------------------------------------------------
// WMMA fragment load for NT GEMM (A: [M x K] row-major, W: [N x K] row-major).
// 16x32 bf16 tile per fragment; per ISA 7.12.2 the 16-bit A operand is
// row-per-lane with K halves {0-7,16-23} (lanes 0-15) / {8-15,24-31} (16-31).
// B operand mirrors it (row n of W per lane n).
// ---------------------------------------------------------------------------
__device__ __forceinline__ v16bf load_tile_nt(const bf16_t* base, int ld) {
  const int lane = threadIdx.x & 31;
  const int r    = lane & 15;
  const int half = lane >> 4;
  const bf16_t* p = base + (size_t)r * ld + half * 8;
  union { v16bf v; v4f h[2]; } u;
  u.h[0] = *reinterpret_cast<const v4f*>(p);        // K = half*8 .. +7
  u.h[1] = *reinterpret_cast<const v4f*>(p + 16);   // K = half*8+16 .. +23
  return u.v;
}

// Same fragment pattern out of an LDS-staged tile (row stride 64 elements).
__device__ __forceinline__ v16bf load_tile_lds(const bf16_t* base, int ks) {
  const int lane = threadIdx.x & 31;
  const int r    = lane & 15;
  const int half = lane >> 4;
  const bf16_t* p = base + (size_t)r * 64 + ks + half * 8;
  union { v16bf v; v4f h[2]; } u;
  u.h[0] = *reinterpret_cast<const v4f*>(p);
  u.h[1] = *reinterpret_cast<const v4f*>(p + 16);
  return u.v;
}

__device__ __forceinline__ v8f wmma_bf16(v16bf a, v16bf b, v8f c) {
  return __builtin_amdgcn_wmma_f32_16x16x32_bf16(false, a, false, b,
                                                 (short)0, c, false, false);
}

// Epilogues:
//  EPI=0: f32 out,  Cf = acc*scale + bias[col]          (scores, out-proj)
//  EPI=1: bf16 out, Cb = bf16(acc + bias[col])          (QKV, attn*V)
//  EPI=2: bf16 out, Cb = bf16(gelu(acc+bias)*wdense)    (MoE up-proj, scaled)
// NT = number of 16-wide N tiles per wave (2 -> 128x64 block, 4 -> 128x128).
template <int EPI, int NT>
__device__ __forceinline__ void gemm_core(const bf16_t* __restrict__ A, int lda,
                                          const bf16_t* __restrict__ W, int ldw,
                                          const float* __restrict__ bias,
                                          float scale,
                                          float* __restrict__ Cf,
                                          bf16_t* __restrict__ Cb, int ldc,
                                          int K,
                                          const float* __restrict__ wdense,
                                          int expert) {
  // 8 waves: 4 (M) x 2 (N). Wave tile: 32 x (NT*16).
  const int wave = threadIdx.x >> 5;
  const int m0 = blockIdx.x * 128 + (wave & 3) * 32;
  const int n0 = blockIdx.y * (NT * 32) + (wave >> 2) * (NT * 16);

  v8f z = {0.f, 0.f, 0.f, 0.f, 0.f, 0.f, 0.f, 0.f};
  v8f acc[2][NT];
  for (int i = 0; i < 2; ++i)
    for (int j = 0; j < NT; ++j) acc[i][j] = z;

  const bf16_t* Ap0 = A + (size_t)m0 * lda;
  const bf16_t* Ap1 = A + (size_t)(m0 + 16) * lda;
  const bf16_t* Wp[NT];
  for (int j = 0; j < NT; ++j) Wp[j] = W + (size_t)(n0 + j * 16) * ldw;

  for (int k = 0; k < K; k += 32) {
    __builtin_prefetch(Ap0 + k + 1024, 0, 1);  // global_prefetch_b8
    v16bf a0 = load_tile_nt(Ap0 + k, lda);
    v16bf a1 = load_tile_nt(Ap1 + k, lda);
    for (int j = 0; j < NT; ++j) {
      v16bf bj = load_tile_nt(Wp[j] + k, ldw);
      acc[0][j] = wmma_bf16(a0, bj, acc[0][j]);
      acc[1][j] = wmma_bf16(a1, bj, acc[1][j]);
    }
  }

  // C/D layout: lane n holds column n; lanes 16-31 hold rows 8-15.
  const int lane = threadIdx.x & 31;
  const int nloc = lane & 15;
  const int half = lane >> 4;
  for (int i = 0; i < 2; ++i)
    for (int j = 0; j < NT; ++j) {
      const int colg = n0 + j * 16 + nloc;
      const float bv = bias ? bias[colg] : 0.f;
      for (int r = 0; r < 8; ++r) {
        const int rowg = m0 + i * 16 + half * 8 + r;
        const float v = acc[i][j][r];
        if constexpr (EPI == 0) {
          Cf[(size_t)rowg * ldc + colg] = v * scale + bv;
        } else if constexpr (EPI == 1) {
          Cb[(size_t)rowg * ldc + colg] = (bf16_t)(v + bv);
        } else {  // EPI == 2: exact gelu, scaled by per-token expert weight
          const float g  = v + bv;
          const float ge = 0.5f * g * (1.f + erff(g * 0.70710678118654752f));
          Cb[(size_t)rowg * ldc + colg] =
              (bf16_t)(ge * wdense[rowg * NE + expert]);
        }
      }
    }
}

// Generic batched NT GEMM (z-dim selects expert / batch slice via strides).
template <int EPI, int NT>
__global__ __launch_bounds__(256) void k_gemm(
    const bf16_t* __restrict__ A, int lda, long sA,
    const bf16_t* __restrict__ W, int ldw, long sW,
    const float* __restrict__ bias, long sBias, float scale,
    float* __restrict__ Cf, bf16_t* __restrict__ Cb, int ldc, long sC,
    int K, const float* __restrict__ wdense) {
  const int zc = blockIdx.z;
  A += (size_t)zc * sA;
  W += (size_t)zc * sW;
  if (bias) bias += (size_t)zc * sBias;
  if (Cf) Cf += (size_t)zc * sC;
  if (Cb) Cb += (size_t)zc * sC;
  gemm_core<EPI, NT>(A, lda, W, ldw, bias, scale, Cf, Cb, ldc, K, wdense, zc);
}

// scores[bh] = (Q_bh @ K_bh^T) / sqrt(HD), f32 out. z = b*16 + h.
__global__ __launch_bounds__(256) void k_scores(const bf16_t* __restrict__ qkv,
                                                float* __restrict__ scores) {
  const int zc = blockIdx.z, b = zc >> 4, h = zc & 15;
  const bf16_t* A = qkv + (size_t)b * SEQ * (3 * DIM) + h * HDIM;          // Q
  const bf16_t* W = qkv + (size_t)b * SEQ * (3 * DIM) + DIM + h * HDIM;    // K
  float* Cf = scores + (size_t)zc * SEQ * SEQ;
  gemm_core<0, 4>(A, 3 * DIM, W, 3 * DIM, nullptr, 0.125f, Cf, nullptr, SEQ,
                  HDIM, nullptr, 0);
}

// O[bh] = P_bh @ V_bh  (vT is V^T per head: [HDIM x SEQ]) -> bf16 o buffer.
__global__ __launch_bounds__(256) void k_attn_o(const bf16_t* __restrict__ P,
                                                const bf16_t* __restrict__ vT,
                                                bf16_t* __restrict__ ob) {
  const int zc = blockIdx.z, b = zc >> 4, h = zc & 15;
  const bf16_t* A = P + (size_t)zc * SEQ * SEQ;
  const bf16_t* W = vT + (size_t)zc * HDIM * SEQ;
  bf16_t* Cb = ob + (size_t)b * SEQ * DIM + h * HDIM;
  gemm_core<1, 2>(A, SEQ, W, SEQ, nullptr, 1.f, nullptr, Cb, DIM, SEQ,
                  nullptr, 0);
}

// ---------------------------------------------------------------------------
// MoE down-proj, accumulated over all 8 experts in-register, with the weight
// (B) tiles double-buffered through LDS via global_load_async_to_lds_b128.
// Block tile 128(M) x 128(N); K chunks of 64 (16 KB of weights per chunk).
// out[t][d] = sum_e Hs_e[t] @ w2_e[d]^T + sum_e wdense[t][e]*b2[e][d]
// ---------------------------------------------------------------------------
__global__ __launch_bounds__(256) void k_moe_out(
    const bf16_t* __restrict__ Hs, const bf16_t* __restrict__ W2,
    const float* __restrict__ b2, const float* __restrict__ wdense,
    float* __restrict__ Out) {
  __shared__ bf16_t Bsh[2][128 * 64];  // 2 x 16 KB

  const int wave = threadIdx.x >> 5;
  const int wc   = wave >> 2;                 // wave column (0..1)
  const int m0   = blockIdx.x * 128 + (wave & 3) * 32;
  const int nblk = blockIdx.y * 128;

  v8f z = {0.f, 0.f, 0.f, 0.f, 0.f, 0.f, 0.f, 0.f};
  v8f acc[2][4];
  for (int i = 0; i < 2; ++i)
    for (int j = 0; j < 4; ++j) acc[i][j] = z;

  constexpr int CHUNKS = NE * (FF / 64);  // 512

  // Stage one 128x64 bf16 weight chunk into Bsh[buf] (4 async b128 per lane).
  auto stage = [&](int s, int buf) {
    const int e  = s >> 6;
    const int kc = (s & 63) * 64;
    const bf16_t* Wrow0 = W2 + (size_t)e * DIM * FF + (size_t)nblk * FF + kc;
    const uint32_t lbase = (uint32_t)(uintptr_t)(&Bsh[buf][0]);
    for (int q = 0; q < 4; ++q) {
      const int idx = threadIdx.x + q * 256;  // 0..1023
      const int nl = idx >> 3, seg = idx & 7; // row in tile, 16B segment
      async_load_b128(lbase + (uint32_t)idx * 16,
                      Wrow0 + (size_t)nl * FF + seg * 8);
    }
  };

  stage(0, 0);
  for (int s = 0; s < CHUNKS; ++s) {
    const int buf = s & 1;
    if (s + 1 < CHUNKS) {
      stage(s + 1, buf ^ 1);
      wait_asynccnt<4>();  // in-order: chunk s fully landed in LDS
    } else {
      wait_asynccnt<0>();
    }
    __syncthreads();

    const int e  = s >> 6;
    const int kc = (s & 63) * 64;
    const bf16_t* Ap0 = Hs + (size_t)e * TOK * FF + (size_t)m0 * FF + kc;
    const bf16_t* Ap1 = Ap0 + (size_t)16 * FF;
    for (int ks = 0; ks < 64; ks += 32) {
      __builtin_prefetch(Ap0 + ks + 1024, 0, 1);
      v16bf a0 = load_tile_nt(Ap0 + ks, FF);
      v16bf a1 = load_tile_nt(Ap1 + ks, FF);
      for (int j = 0; j < 4; ++j) {
        const bf16_t* bb = &Bsh[buf][(wc * 64 + j * 16) * 64];
        v16bf bj = load_tile_lds(bb, ks);
        acc[0][j] = wmma_bf16(a0, bj, acc[0][j]);
        acc[1][j] = wmma_bf16(a1, bj, acc[1][j]);
      }
    }
    __syncthreads();  // protect Bsh[buf] before it is re-staged
  }

  const int lane = threadIdx.x & 31;
  const int nloc = lane & 15;
  const int half = lane >> 4;
  for (int i = 0; i < 2; ++i)
    for (int j = 0; j < 4; ++j) {
      const int colg = nblk + wc * 64 + j * 16 + nloc;
      for (int r = 0; r < 8; ++r) {
        const int rowg = m0 + i * 16 + half * 8 + r;
        float bv = 0.f;
        for (int e = 0; e < NE; ++e)
          bv += wdense[rowg * NE + e] * b2[e * DIM + colg];
        Out[(size_t)rowg * DIM + colg] = acc[i][j][r] + bv;
      }
    }
}

// ---------------------------------------------------------------------------
// Support kernels (fp32 numerics)
// ---------------------------------------------------------------------------
__global__ void k_cvt(const float* __restrict__ in, bf16_t* __restrict__ out,
                      size_t n) {
  size_t i = (size_t)blockIdx.x * blockDim.x + threadIdx.x;
  const size_t stride = (size_t)gridDim.x * blockDim.x;
  for (; i < n; i += stride) out[i] = (bf16_t)in[i];
}

// vT[(b*16+h)][col][key] = qkv[b*S+key][2D + h*64 + col]
__global__ void k_make_vT(const bf16_t* __restrict__ qkv,
                          bf16_t* __restrict__ vT) {
  const size_t i = (size_t)blockIdx.x * blockDim.x + threadIdx.x;  // 2M threads
  const int key = i & 1023;
  const int col = (i >> 10) & 63;
  const int bh  = i >> 16;
  const int b = bh >> 4, h = bh & 15;
  vT[i] = qkv[(size_t)(b * SEQ + key) * (3 * DIM) + 2 * DIM + h * HDIM + col];
}

__global__ __launch_bounds__(256) void k_softmax(const float* __restrict__ Sc,
                                                 bf16_t* __restrict__ P) {
  const size_t row = blockIdx.x;
  const float* x = Sc + row * SEQ;
  bf16_t* p = P + row * SEQ;
  __shared__ float red[256];
  const int t = threadIdx.x;
  float m = -1e30f;
  for (int c = t; c < SEQ; c += 256) m = fmaxf(m, x[c]);
  red[t] = m; __syncthreads();
  for (int s = 128; s > 0; s >>= 1) {
    if (t < s) red[t] = fmaxf(red[t], red[t + s]);
    __syncthreads();
  }
  m = red[0]; __syncthreads();
  float vals[4], sum = 0.f;
  for (int i = 0; i < 4; ++i) {
    vals[i] = expf(x[t + i * 256] - m);
    sum += vals[i];
  }
  red[t] = sum; __syncthreads();
  for (int s = 128; s > 0; s >>= 1) {
    if (t < s) red[t] += red[t + s];
    __syncthreads();
  }
  const float inv = 1.f / red[0];
  for (int i = 0; i < 4; ++i) p[t + i * 256] = (bf16_t)(vals[i] * inv);
}

// out = LN(a + b) * g + beta; optional bf16 mirror for downstream WMMA.
__global__ __launch_bounds__(256) void k_ln(const float* __restrict__ a,
                                            const float* __restrict__ b,
                                            const float* __restrict__ g,
                                            const float* __restrict__ beta,
                                            float* __restrict__ outf,
                                            bf16_t* __restrict__ outb) {
  const size_t tk = blockIdx.x;
  const float* pa = a + tk * DIM;
  const float* pb = b + tk * DIM;
  __shared__ float red[256];
  const int t = threadIdx.x;
  float x[4], s = 0.f;
  for (int i = 0; i < 4; ++i) {
    x[i] = pa[t + i * 256] + pb[t + i * 256];
    s += x[i];
  }
  red[t] = s; __syncthreads();
  for (int st = 128; st > 0; st >>= 1) {
    if (t < st) red[t] += red[t + st];
    __syncthreads();
  }
  const float mu = red[0] * (1.f / DIM);
  __syncthreads();
  float v = 0.f;
  for (int i = 0; i < 4; ++i) { const float d = x[i] - mu; v += d * d; }
  red[t] = v; __syncthreads();
  for (int st = 128; st > 0; st >>= 1) {
    if (t < st) red[t] += red[t + st];
    __syncthreads();
  }
  const float rstd = rsqrtf(red[0] * (1.f / DIM) + 1e-5f);
  for (int i = 0; i < 4; ++i) {
    const int c = t + i * 256;
    const float y = (x[i] - mu) * rstd * g[c] + beta[c];
    outf[tk * DIM + c] = y;
    if (outb) outb[tk * DIM + c] = (bf16_t)y;
  }
}

// Per-token gate: logits (8 dots of length 1024), full softmax probs (for aux
// usage), top-2 renormalized weights -> wdense. One wave per token.
__global__ __launch_bounds__(256) void k_gate(const float* __restrict__ x,
                                              const float* __restrict__ gw,
                                              const float* __restrict__ gb,
                                              float* __restrict__ wdense,
                                              float* __restrict__ probs8) {
  const int tok = blockIdx.x * 8 + (threadIdx.x >> 5);
  const int lane = threadIdx.x & 31;
  float lg[NE];
  for (int e = 0; e < NE; ++e) lg[e] = 0.f;
  const float* xr = x + (size_t)tok * DIM;
  for (int d = lane; d < DIM; d += 32) {
    const float xv = xr[d];
    for (int e = 0; e < NE; ++e) lg[e] += xv * gw[e * DIM + d];
  }
  for (int e = 0; e < NE; ++e) {
    for (int off = 16; off > 0; off >>= 1) lg[e] += __shfl_xor(lg[e], off, 32);
    lg[e] += gb[e];
  }
  // full softmax (usage / aux loss)
  float mx = lg[0];
  for (int e = 1; e < NE; ++e) mx = fmaxf(mx, lg[e]);
  float pr[NE], s = 0.f;
  for (int e = 0; e < NE; ++e) { pr[e] = expf(lg[e] - mx); s += pr[e]; }
  const float invs = 1.f / s;
  // top-2 (first index wins ties, matching lax.top_k)
  int i1 = 0;
  for (int e = 1; e < NE; ++e) if (lg[e] > lg[i1]) i1 = e;
  int i2 = (i1 == 0) ? 1 : 0;
  for (int e = 0; e < NE; ++e) if (e != i1 && lg[e] > lg[i2]) i2 = e;
  const float e2 = expf(lg[i2] - lg[i1]);
  const float p1 = 1.f / (1.f + e2);
  const float p2 = e2 / (1.f + e2);
  if (lane < NE) {
    float myp = 0.f, myw = 0.f;
    for (int e = 0; e < NE; ++e)
      if (lane == e) { myp = pr[e] * invs; myw = (e == i1) ? p1 : ((e == i2) ? p2 : 0.f); }
    probs8[tok * NE + lane] = myp;
    wdense[tok * NE + lane] = myw;
  }
}

// aux = E * sum_e (mean_t probs[t][e])^2  (deterministic serial reduction)
__global__ void k_aux(const float* __restrict__ probs8, float* __restrict__ out) {
  __shared__ float us[NE];
  const int t = threadIdx.x;
  if (t < NE) {
    float s = 0.f;
    for (int i = 0; i < TOK; ++i) s += probs8[i * NE + t];
    us[t] = s * (1.f / TOK);
  }
  __syncthreads();
  if (t == 0) {
    float s = 0.f;
    for (int e = 0; e < NE; ++e) s += us[e] * us[e];
    out[0] = (float)NE * s;
  }
}

// ---------------------------------------------------------------------------
extern "C" void kernel_launch(void* const* d_in, const int* in_sizes, int n_in,
                              void* d_out, int out_size, void* d_ws,
                              size_t ws_size, hipStream_t stream) {
  const float* src  = (const float*)d_in[0];
  const float* ipw  = (const float*)d_in[1];
  const float* ipb  = (const float*)d_in[2];
  const float* outw = (const float*)d_in[3];
  const float* outbv= (const float*)d_in[4];
  const float* gw   = (const float*)d_in[5];
  const float* gb   = (const float*)d_in[6];
  const float* w1   = (const float*)d_in[7];
  const float* b1   = (const float*)d_in[8];
  const float* w2   = (const float*)d_in[9];
  const float* b2   = (const float*)d_in[10];
  const float* ln1g = (const float*)d_in[11];
  const float* ln1b = (const float*)d_in[12];
  const float* ln2g = (const float*)d_in[13];
  const float* ln2b = (const float*)d_in[14];
  float* out = (float*)d_out;

  // Workspace carve-up (256B aligned).
  char* base = (char*)d_ws;
  size_t off = 0;
  auto carve = [&](size_t bytes) {
    char* p = base + off;
    off += (bytes + 255) & ~(size_t)255;
    return p;
  };
  bf16_t* src_b   = (bf16_t*)carve((size_t)TOK * DIM * 2);          // 4 MB
  bf16_t* wqkv_b  = (bf16_t*)carve((size_t)3 * DIM * DIM * 2);      // 6 MB
  bf16_t* wout_b  = (bf16_t*)carve((size_t)DIM * DIM * 2);          // 2 MB
  bf16_t* w1_b    = (bf16_t*)carve((size_t)NE * FF * DIM * 2);      // 64 MB
  bf16_t* w2_b    = (bf16_t*)carve((size_t)NE * DIM * FF * 2);      // 64 MB
  bf16_t* qkv_b   = (bf16_t*)carve((size_t)TOK * 3 * DIM * 2);      // 12 MB
  bf16_t* vT      = (bf16_t*)carve((size_t)32 * HDIM * SEQ * 2);    // 4 MB
  bf16_t* Pbuf    = (bf16_t*)carve((size_t)32 * SEQ * SEQ * 2);     // 64 MB
  bf16_t* o_b     = (bf16_t*)carve((size_t)TOK * DIM * 2);          // 4 MB
  float*  attnprj = (float*)carve((size_t)TOK * DIM * 4);           // 8 MB
  float*  x1      = (float*)carve((size_t)TOK * DIM * 4);           // 8 MB
  bf16_t* x1_b    = (bf16_t*)carve((size_t)TOK * DIM * 2);          // 4 MB
  float*  moeout  = (float*)carve((size_t)TOK * DIM * 4);           // 8 MB
  float*  wdense  = (float*)carve((size_t)TOK * NE * 4);
  float*  probs8  = (float*)carve((size_t)TOK * NE * 4);
  // scores (f32, 128 MB) aliases MoE hidden (bf16, 128 MB): scores is dead
  // after softmax; Hs written long after.
  char*   big     = carve((size_t)32 * SEQ * SEQ * 4);              // 128 MB
  float*  scores  = (float*)big;
  bf16_t* Hs      = (bf16_t*)big;

  // 1) fp32 -> bf16 conversions (one-time, bandwidth-trivial).
  k_cvt<<<2048, 256, 0, stream>>>(src, src_b, (size_t)TOK * DIM);
  k_cvt<<<2048, 256, 0, stream>>>(ipw, wqkv_b, (size_t)3 * DIM * DIM);
  k_cvt<<<2048, 256, 0, stream>>>(outw, wout_b, (size_t)DIM * DIM);
  k_cvt<<<8192, 256, 0, stream>>>(w1, w1_b, (size_t)NE * FF * DIM);
  k_cvt<<<8192, 256, 0, stream>>>(w2, w2_b, (size_t)NE * DIM * FF);

  // 2) QKV projection: [2048x1024] @ [3072x1024]^T + b -> bf16 [2048x3072]
  k_gemm<1, 4><<<dim3(TOK / 128, 3 * DIM / 128, 1), 256, 0, stream>>>(
      src_b, DIM, 0L, wqkv_b, DIM, 0L, ipb, 0L, 1.f,
      nullptr, qkv_b, 3 * DIM, 0L, DIM, nullptr);

  // 3) V^T per head for the attn*V NT GEMM.
  k_make_vT<<<8192, 256, 0, stream>>>(qkv_b, vT);

  // 4) scores = QK^T/8 (32 heads), softmax -> bf16 P, O = P@V.
  k_scores<<<dim3(SEQ / 128, SEQ / 128, 32), 256, 0, stream>>>(qkv_b, scores);
  k_softmax<<<32 * SEQ, 256, 0, stream>>>(scores, Pbuf);
  k_attn_o<<<dim3(SEQ / 128, HDIM / 64, 32), 256, 0, stream>>>(Pbuf, vT, o_b);

  // 5) out-proj (f32) then LN1(src + attn) -> x1 (f32 + bf16 mirror).
  k_gemm<0, 4><<<dim3(TOK / 128, DIM / 128, 1), 256, 0, stream>>>(
      o_b, DIM, 0L, wout_b, DIM, 0L, outbv, 0L, 1.f,
      attnprj, nullptr, DIM, 0L, DIM, nullptr);
  k_ln<<<TOK, 256, 0, stream>>>(src, attnprj, ln1g, ln1b, x1, x1_b);

  // 6) gating: dense softmax probs (aux) + top-2 weights.
  k_gate<<<TOK / 8, 256, 0, stream>>>(x1, gw, gb, wdense, probs8);

  // 7) MoE up-proj, all experts dense: Hs_e = gelu(x1@w1_e^T+b1_e)*wdense_e.
  k_gemm<2, 4><<<dim3(TOK / 128, FF / 128, NE), 256, 0, stream>>>(
      x1_b, DIM, 0L, w1_b, DIM, (long)FF * DIM, b1, (long)FF, 1.f,
      nullptr, Hs, FF, (long)TOK * FF, DIM, wdense);

  // 8) MoE down-proj: async-LDS staged weights, 8 experts in-register.
  k_moe_out<<<dim3(TOK / 128, DIM / 128, 1), 256, 0, stream>>>(
      Hs, w2_b, b2, wdense, moeout);

  // 9) LN2(x1 + moe) -> d_out; aux loss scalar appended.
  k_ln<<<TOK, 256, 0, stream>>>(x1, moeout, ln2g, ln2b, out, nullptr);
  k_aux<<<1, 32, 0, stream>>>(probs8, out + (size_t)TOK * DIM);

  (void)in_sizes; (void)n_in; (void)out_size; (void)ws_size;
}
